// GCNEncoder_40776419508781
// MI455X (gfx1250) — matrix operand order repaired and checked by reading, not compiled, not run on hardware
//
#include <hip/hip_runtime.h>
#include <cstddef>

// ---------------- types ----------------
typedef __attribute__((ext_vector_type(16))) __bf16          v16bf;
typedef __attribute__((ext_vector_type(4)))  __bf16          bfv4;
typedef __attribute__((ext_vector_type(8)))  float           v8f;
typedef __attribute__((ext_vector_type(4)))  float           fv4;
typedef __attribute__((ext_vector_type(4)))  unsigned int    uiv4;
typedef __attribute__((ext_vector_type(4)))  unsigned short  usv4;

union Frag16 {            // one WMMA bf16 operand: 16 bf16 per lane = 8 VGPRs
  v16bf v;
  uiv4  q[2];
  usv4  w[4];
  unsigned short u[16];
};

// packed f32x4 -> bf16x4 (RNE) -- lets clang emit v_cvt_pk_bf16_f32
__device__ __forceinline__ usv4 cvt4(fv4 v) {
  union { bfv4 b; usv4 u; } c;
  c.b = __builtin_convertvector(v, bfv4);
  return c.u;
}
__device__ __forceinline__ unsigned short f2bf(float f) {
  union { __bf16 b; unsigned short u; } c;
  c.b = (__bf16)f;
  return c.u;
}
__device__ __forceinline__ float bf2f(unsigned short u) {
  return __uint_as_float(((unsigned int)u) << 16);
}
__device__ __forceinline__ v8f wmma_bf16(const Frag16& a, const Frag16& b, v8f c) {
  return __builtin_amdgcn_wmma_f32_16x16x32_bf16(false, a.v, false, b.v,
                                                 (short)0, c, false, false);
}

#define NROWS 8192

// -------- CDNA5 Tensor Data Mover feed (guarded so compile can't regress) --------
#if defined(__AMDGCN__) && __has_builtin(__builtin_amdgcn_tensor_load_to_lds) && __has_builtin(__builtin_amdgcn_s_wait_tensorcnt)
#define CDNA5_TDM 1
#else
#define CDNA5_TDM 0
#endif

#if CDNA5_TDM
typedef __attribute__((ext_vector_type(4))) unsigned int u32x4;
typedef __attribute__((ext_vector_type(4))) int          i32x4;
typedef __attribute__((ext_vector_type(8))) int          i32x8;

// 32-bit LDS byte address of a shared-memory pointer (AS3 pointers are 32-bit)
__device__ __forceinline__ unsigned lds_off(const void* p) {
  return (unsigned)(unsigned long long)(__attribute__((address_space(3))) const void*)p;
}

// TDM: load one 64(rows) x 32(bf16) tile from a row-major [8192 x 8192] bf16
// tensor into LDS, inserting 16B of padding after each 64B row (LDT=40 shorts).
__device__ __forceinline__ void tdm_load_tile_bf16(const void* gtile, unsigned ldsaddr) {
  const unsigned long long ga = (unsigned long long)gtile;
  u32x4 g0;
  g0.x = 1u;                                            // count=1 (valid user D#)
  g0.y = ldsaddr;                                       // lds_addr  (bits 63:32)
  g0.z = (unsigned)(ga & 0xFFFFFFFFu);                  // global_addr[31:0]
  g0.w = (unsigned)((ga >> 32) & 0x01FFFFFFu)           // global_addr[56:32]
       | (2u << 30);                                    // type = 2 ("image")
  i32x8 g1;
  g1[0] = (1 << 16)                                     // data_size = 1 -> 2 bytes
        | (1 << 20)                                     // pad_enable
        | (3 << 22)                                     // pad_interval: every 16 DWORDs (64B row)
        | (3 << 25);                                    // pad_amount: 4 DWORDs (16B)
  g1[1] = (int)(8192u << 16);                           // tensor_dim0[15:0] in [31:16]
  g1[2] = (int)(8192u << 16);                           // tensor_dim0 hi=0 | tensor_dim1[15:0]
  g1[3] = (int)(32u << 16);                             // tensor_dim1 hi=0 | tile_dim0 = 32
  g1[4] = 64;                                           // tile_dim1 = 64 | tile_dim2 = 0
  g1[5] = 8192;                                         // tensor_dim0_stride[31:0]
  g1[6] = 0;                                            // stride hi | tensor_dim1_stride lo
  g1[7] = 0;
  const i32x4 z4 = {0, 0, 0, 0};
#if __clang_major__ >= 23
  const i32x8 z8 = {0, 0, 0, 0, 0, 0, 0, 0};
  __builtin_amdgcn_tensor_load_to_lds(g0, g1, z4, z4, z8, 0);
#else
  __builtin_amdgcn_tensor_load_to_lds(g0, g1, z4, z4, 0);
#endif
}
#endif  // CDNA5_TDM

// ---------------- degree / scaling (+ optional bf16 copy of A) ----------------
template<bool WRITE_BF>
__global__ void __launch_bounds__(256)
deg_kernel(const float* __restrict__ A, float* __restrict__ s,
           unsigned short* __restrict__ Abf) {
  __shared__ float red[256];
  const int row = blockIdx.x;
  const int t   = threadIdx.x;
  const float* arow = A + (size_t)row * NROWS;
  float sum = 0.f;
  #pragma unroll
  for (int p = 0; p < 8; ++p) {
    const int j = p * 1024 + t * 4;
    fv4 v = *(const fv4*)(arow + j);
    sum += (v.x + v.y) + (v.z + v.w);
    if constexpr (WRITE_BF)
      *(usv4*)(Abf + (size_t)row * NROWS + j) = cvt4(v);
  }
  red[t] = sum;
  __syncthreads();
  for (int off = 128; off > 0; off >>= 1) {
    if (t < off) red[t] += red[t + off];
    __syncthreads();
  }
  if (t == 0) s[row] = rsqrtf(red[0] + 1.0f);   // deg = rowsum(A) + 1 (the +I)
}

// ---------------- small GEMM: Out = bf16( s[i] * (In @ W^T + bias) ) ----------------
template<bool IN_BF16>
__global__ void __launch_bounds__(256)
gemm_kernel(const void* __restrict__ inp, const float* __restrict__ W,
            const float* __restrict__ bias, const float* __restrict__ s,
            unsigned short* __restrict__ outp, int F, int K) {
  const int tid  = threadIdx.x;
  const int lane = tid & 31;
  const int wave = tid >> 5;                    // 0..7
  const int tilesK = K >> 4;
  const int g  = blockIdx.x * 8 + wave;
  const int i0 = (g / tilesK) << 4;
  const int k0 = (g % tilesK) << 4;
  const int m  = lane & 15;
  const int h  = lane >> 4;

  v8f acc = {0.f,0.f,0.f,0.f,0.f,0.f,0.f,0.f};

  for (int f0 = 0; f0 < F; f0 += 32) {
    Frag16 a, b;
    // A operand: row (i0+m), K runs [8h..8h+7] and [16+8h..16+8h+7]
    if constexpr (IN_BF16) {
      const unsigned short* p = (const unsigned short*)inp + (size_t)(i0 + m) * F + f0;
      a.q[0] = *(const uiv4*)(p + 8 * h);
      a.q[1] = *(const uiv4*)(p + 16 + 8 * h);
    } else {
      const float* p = (const float*)inp + (size_t)(i0 + m) * F + f0;
      a.w[0] = cvt4(*(const fv4*)(p + 8 * h));
      a.w[1] = cvt4(*(const fv4*)(p + 8 * h + 4));
      a.w[2] = cvt4(*(const fv4*)(p + 16 + 8 * h));
      a.w[3] = cvt4(*(const fv4*)(p + 16 + 8 * h + 4));
    }
    // B operand: column n = (k0+m) of W^T == row (k0+m) of W; K runs [8h..8h+7] and [16+8h..16+8h+7]
    {
      const float* p = W + (size_t)(k0 + m) * F + f0 + 8 * h;
      b.w[0] = cvt4(*(const fv4*)(p));
      b.w[1] = cvt4(*(const fv4*)(p + 4));
      b.w[2] = cvt4(*(const fv4*)(p + 16));
      b.w[3] = cvt4(*(const fv4*)(p + 20));
    }
    acc = wmma_bf16(a, b, acc);
  }

  const int   kk = k0 + m;
  const float bk = bias[kk];
  #pragma unroll
  for (int r = 0; r < 8; ++r) {
    const int i = i0 + r + 8 * h;               // C/D layout: M = r + 8*(lane/16)
    const float val = (acc[r] + bk) * s[i];
    outp[(size_t)i * K + kk] = f2bf(val);
  }
}

// ---------------- aggregation: Y = post( s[i] * ((A+I) @ M)[i][c] ) ----------------
// 64(i) x 64(c) block tile, 128 threads = 4 waves (2x2 wave grid, 32x32 each).
// Double-buffered LDS pipeline over j (BK=32). When A is the bf16 L2-resident copy,
// the A-tile feed is a single TDM tensor_load_to_lds per step (TENSORcnt-tracked);
// otherwise A is converted on the fly through registers.
template<typename AT, bool RELU, bool OUT_BF16>
__global__ void __launch_bounds__(128)
agg_kernel(const AT* __restrict__ A, const unsigned short* __restrict__ Mb,
           const float* __restrict__ s, void* __restrict__ outp, int C) {
  constexpr int BK = 32, LDT = BK + 8;          // 64B row + 16B pad (TDM pad matches)
  __shared__ unsigned short As[2][64][LDT];     // [buf][i][j]
  __shared__ unsigned short Bs[2][64][LDT];     // [buf][c][j]  (M tile, transposed)

#if CDNA5_TDM
  constexpr bool kTDM = (sizeof(AT) == 2);      // bf16 A: pure 2D copy -> TDM
#else
  constexpr bool kTDM = false;
#endif

  const int tid  = threadIdx.x;
  const int lane = tid & 31;
  const int wave = tid >> 5;                    // 0..3
  const int wm   = wave & 1;
  const int wn   = wave >> 1;
  const int i0   = blockIdx.x * 64;
  const int c0   = blockIdx.y * 64;
  const int m    = lane & 15;
  const int h    = lane >> 4;

  v8f acc[2][2];
  #pragma unroll
  for (int fa = 0; fa < 2; ++fa)
    #pragma unroll
    for (int fb = 0; fb < 2; ++fb)
      acc[fa][fb] = (v8f){0.f,0.f,0.f,0.f,0.f,0.f,0.f,0.f};

  const int arow = tid >> 3;                    // 0..15
  const int acol = (tid & 7) * 4;               // 0..28
  const int bj   = tid >> 4;                    // 0..7
  const int bc   = (tid & 15) * 4;              // 0..60

  usv4 aregs[4], bregs[4];

  auto issue_or_load_A = [&](int j0, int buf) {
    if constexpr (kTDM) {
#if CDNA5_TDM
      if (wave == 0)                            // one DMA per workgroup per tile
        tdm_load_tile_bf16((const unsigned short*)A + (size_t)i0 * NROWS + j0,
                           lds_off(&As[buf][0][0]));
#endif
    } else {
      #pragma unroll
      for (int p = 0; p < 4; ++p) {
        const int r = arow + p * 16;
        if constexpr (sizeof(AT) == 4) {
          const float* gp = (const float*)A + (size_t)(i0 + r) * NROWS + j0 + acol;
          aregs[p] = cvt4(*(const fv4*)gp);
          if (j0 + 8 * BK < NROWS) __builtin_prefetch(gp + 8 * BK, 0, 0);
        } else {
          aregs[p] = *(const usv4*)((const unsigned short*)A + (size_t)(i0 + r) * NROWS + j0 + acol);
        }
      }
    }
  };
  auto store_A = [&](int buf) {
    if constexpr (!kTDM) {
      #pragma unroll
      for (int p = 0; p < 4; ++p)
        *(usv4*)&As[buf][arow + p * 16][acol] = aregs[p];
    }
  };
  auto load_B = [&](int j0) {
    #pragma unroll
    for (int p = 0; p < 4; ++p)
      bregs[p] = *(const usv4*)(Mb + (size_t)(j0 + bj + p * 8) * C + c0 + bc);
  };
  auto store_B = [&](int buf) {
    #pragma unroll
    for (int p = 0; p < 4; ++p) {
      const int j = bj + p * 8;
      Bs[buf][bc + 0][j] = bregs[p].x; Bs[buf][bc + 1][j] = bregs[p].y;
      Bs[buf][bc + 2][j] = bregs[p].z; Bs[buf][bc + 3][j] = bregs[p].w;
    }
  };
  auto wait_feed = [&]() {
    if constexpr (kTDM) {
#if CDNA5_TDM
      if (wave == 0) __builtin_amdgcn_s_wait_tensorcnt(0);
#endif
    }
  };

  // ---- prologue: fill buffer 0 ----
  issue_or_load_A(0, 0);
  load_B(0);
  store_A(0);
  store_B(0);
  wait_feed();
  __syncthreads();

  constexpr int NSTEP = NROWS / BK;
  for (int k = 0; k < NSTEP; ++k) {
    const int cur = k & 1, nxt = cur ^ 1;
    const bool more = (k + 1 < NSTEP);
    if (more) {                                  // start next tile's feed
      issue_or_load_A((k + 1) * BK, nxt);
      load_B((k + 1) * BK);
    }
    // ---- fragments + 4 WMMAs from current buffer ----
    Frag16 a0, a1, b0, b1;
    {
      const unsigned short* pa0 = &As[cur][wm * 32 + m][0];
      a0.q[0] = *(const uiv4*)(pa0 + 8 * h);
      a0.q[1] = *(const uiv4*)(pa0 + 16 + 8 * h);
      const unsigned short* pa1 = &As[cur][wm * 32 + 16 + m][0];
      a1.q[0] = *(const uiv4*)(pa1 + 8 * h);
      a1.q[1] = *(const uiv4*)(pa1 + 16 + 8 * h);
      const unsigned short* pb0 = &Bs[cur][wn * 32 + m][0];
      b0.q[0] = *(const uiv4*)(pb0 + 8 * h);
      b0.q[1] = *(const uiv4*)(pb0 + 16 + 8 * h);
      const unsigned short* pb1 = &Bs[cur][wn * 32 + 16 + m][0];
      b1.q[0] = *(const uiv4*)(pb1 + 8 * h);
      b1.q[1] = *(const uiv4*)(pb1 + 16 + 8 * h);
    }
    acc[0][0] = wmma_bf16(a0, b0, acc[0][0]);
    acc[0][1] = wmma_bf16(a0, b1, acc[0][1]);
    acc[1][0] = wmma_bf16(a1, b0, acc[1][0]);
    acc[1][1] = wmma_bf16(a1, b1, acc[1][1]);

    if (more) {
      store_A(nxt);
      store_B(nxt);
      wait_feed();                               // next tile's TDM complete (wave 0)
    }
    __syncthreads();
  }

  // ---- epilogue: + identity term, * s[i], relu?, store ----
  #pragma unroll
  for (int fm = 0; fm < 2; ++fm) {
    #pragma unroll
    for (int fn = 0; fn < 2; ++fn) {
      #pragma unroll
      for (int r = 0; r < 8; ++r) {
        const int i = i0 + wm * 32 + fm * 16 + r + 8 * h;
        const int c = c0 + wn * 32 + fn * 16 + m;
        float val = acc[fm][fn][r] + bf2f(Mb[(size_t)i * C + c]);  // +I@M
        val *= s[i];
        if constexpr (RELU) val = fmaxf(val, 0.0f);
        if constexpr (OUT_BF16)
          ((unsigned short*)outp)[(size_t)i * C + c] = f2bf(val);
        else
          ((float*)outp)[(size_t)i * C + c] = val;
      }
    }
  }
}

// ---------------- host ----------------
extern "C" void kernel_launch(void* const* d_in, const int* in_sizes, int n_in,
                              void* d_out, int out_size, void* d_ws, size_t ws_size,
                              hipStream_t stream) {
  const float* x  = (const float*)d_in[0];
  const float* A  = (const float*)d_in[1];
  const float* w1 = (const float*)d_in[2];
  const float* b1 = (const float*)d_in[3];
  const float* w2 = (const float*)d_in[4];
  const float* b2 = (const float*)d_in[5];
  float* out = (float*)d_out;

  char* ws = (char*)d_ws;
  float*          sdeg = (float*)ws;                                  // 32 KB
  unsigned short* M1   = (unsigned short*)(ws + (32 << 10));          // 4 MB  (8192x256 bf16)
  unsigned short* H    = (unsigned short*)(ws + (32 << 10) + (4 << 20));   // 4 MB
  unsigned short* M2   = (unsigned short*)(ws + (32 << 10) + (8 << 20));   // 2 MB
  const size_t abase   = (size_t)(32 << 10) + (10 << 20);
  unsigned short* Abf  = (unsigned short*)(ws + abase);
  const size_t abytes  = (size_t)NROWS * NROWS * 2;                   // 128 MB, L2-resident
  const bool useAbf    = ws_size >= abase + abytes;

  // 1) degree + scaling (and bf16 copy of A if workspace allows)
  if (useAbf) deg_kernel<true ><<<NROWS, 256, 0, stream>>>(A, sdeg, Abf);
  else        deg_kernel<false><<<NROWS, 256, 0, stream>>>(A, sdeg, nullptr);

  // 2) M1 = bf16( s * (x @ w1^T + b1) )   tiles = 512*16 = 8192 waves -> 1024 blocks
  gemm_kernel<false><<<1024, 256, 0, stream>>>(x, w1, b1, sdeg, M1, 512, 256);

  // 3) h = relu( s * ((A+I) @ M1) )  -> bf16
  dim3 g1(NROWS / 64, 256 / 64);
  if (useAbf) agg_kernel<unsigned short, true, true><<<g1, 128, 0, stream>>>(Abf, M1, sdeg, H, 256);
  else        agg_kernel<float,          true, true><<<g1, 128, 0, stream>>>(A,   M1, sdeg, H, 256);

  // 4) M2 = bf16( s * (h @ w2^T + b2) )   tiles = 512*8 = 4096 waves -> 512 blocks
  gemm_kernel<true><<<512, 256, 0, stream>>>(H, w2, b2, sdeg, M2, 256, 128);

  // 5) out = s * ((A+I) @ M2)  -> f32
  dim3 g2(NROWS / 64, 128 / 64);
  if (useAbf) agg_kernel<unsigned short, false, false><<<g2, 128, 0, stream>>>(Abf, M2, sdeg, out, 128);
  else        agg_kernel<float,          false, false><<<g2, 128, 0, stream>>>(A,   M2, sdeg, out, 128);
}
